// TopK_9216999817649
// MI455X (gfx1250) — compile-verified, matrix operand order checked by reading, not run on hardware
//
#include <hip/hip_runtime.h>
#include <hip/hip_bf16.h>
#include <math.h>

typedef __attribute__((ext_vector_type(16))) _Float16 v16h;
typedef __attribute__((ext_vector_type(8)))  float    v8f;

#define CENTER_RADIUS 1.5f
#define GEO_PENALTY   1000000.0f
#define FG_PENALTY    100000000.0f
#define NCAND         10
#define MAXG          256

union V16H { v16h v; _Float16 e[16]; };

struct AnchorConst {
    float bxl, bxr, byt, byb, areaB, clsfg, fg, dist, cx, cy;
};
struct GtConst {
    float gxl, gxr, gyt, gyb, gx, gy, areaG;
};

// Branch-free cost/IoU evaluation, identical in both passes (row top-k and
// column finalize) so the lexicographic threshold stays self-consistent.
// clsfg = cls_loss + FG_PENALTY*(1-fg) is precomputed per anchor.
__device__ __forceinline__ void eval_pair(const AnchorConst& an, const GtConst& gt,
                                          float& iou, float& cost) {
    float tlx = fmaxf(gt.gxl, an.bxl);
    float tly = fmaxf(gt.gyt, an.byt);
    float brx = fminf(gt.gxr, an.bxr);
    float bry = fminf(gt.gyb, an.byb);
    float inter = fmaxf(brx - tlx, 0.0f) * fmaxf(bry - tly, 0.0f);
    float den = gt.areaG + an.areaB - inter;
    iou = inter * __builtin_amdgcn_rcpf(den);       // v_rcp_f32, no IEEE div chain
    iou *= an.fg;
    // eager (non-short-circuit) center test: no EXEC branching
    int inc = (fabsf(an.cx - gt.gx) < an.dist) & (fabsf(an.cy - gt.gy) < an.dist);
    cost = an.clsfg + 3.0f * (-__logf(iou + 1e-8f)) + (inc ? 0.0f : GEO_PENALTY);
}

__device__ __forceinline__ AnchorConst loadAnchorPk(const float4* __restrict__ pk0,
                                                    const float4* __restrict__ pk1,
                                                    const float2* __restrict__ pk2,
                                                    int a) {
    float4 p0 = pk0[a];     // global_load_b128
    float4 p1 = pk1[a];     // global_load_b128
    float2 p2 = pk2[a];     // global_load_b64
    AnchorConst an;
    an.bxl = p0.x; an.bxr = p0.y; an.byt = p0.z; an.byb = p0.w;
    an.areaB = p1.x; an.clsfg = p1.y; an.fg = p1.z; an.dist = p1.w;
    an.cx = p2.x; an.cy = p2.y;
    return an;
}

__device__ __forceinline__ void insIou(float* arr, float v) {
    if (v > arr[NCAND - 1]) {
        int p = NCAND - 1;
        while (p > 0 && v > arr[p - 1]) { arr[p] = arr[p - 1]; --p; }
        arr[p] = v;
    }
}

__device__ __forceinline__ void insCost(float* c, int* x, float cv, int iv) {
    if (cv < c[NCAND - 1] || (cv == c[NCAND - 1] && iv < x[NCAND - 1])) {
        int p = NCAND - 1;
        while (p > 0 && (cv < c[p - 1] || (cv == c[p - 1] && iv < x[p - 1]))) {
            c[p] = c[p - 1]; x[p] = x[p - 1]; --p;
        }
        c[p] = cv; x[p] = iv;
    }
}

// ---------------- kernel 0: init scalar tail outputs --------------------
__global__ void kInit(int* tail, int G) {
    tail[0] = 0;      // num_fg accumulator
    tail[1] = G;
}

// ---------------- kernel 1: per-anchor precompute (packed) --------------
__global__ void __launch_bounds__(256)
kAnchorPrep(const float* __restrict__ out7, const float* __restrict__ tgt,
            const float* __restrict__ grid, const float* __restrict__ stride,
            float4* __restrict__ pk0, float4* __restrict__ pk1,
            float2* __restrict__ pk2, int A, int G) {
    __shared__ float sgx[MAXG], sgy[MAXG];
    int tid = threadIdx.x;
    for (int i = tid; i < G; i += blockDim.x) {
        sgx[i] = tgt[i * 5 + 1];
        sgy[i] = tgt[i * 5 + 2];
    }
    __syncthreads();
    int a = blockIdx.x * blockDim.x + tid;
    if (a >= A) return;
    float st = stride[a];
    float cx = (grid[a * 2 + 0] + 0.5f) * st;
    float cy = (grid[a * 2 + 1] + 0.5f) * st;
    float dist = st * CENTER_RADIUS;
    int fg = 0;
    for (int g = 0; g < G; ++g) {
        fg |= (fabsf(cx - sgx[g]) < dist) & (fabsf(cy - sgy[g]) < dist);
    }
    float bx = out7[a * 7 + 0], by = out7[a * 7 + 1];
    float bw = out7[a * 7 + 2], bh = out7[a * 7 + 3];
    float o4 = out7[a * 7 + 4], o5 = out7[a * 7 + 5], o6 = out7[a * 7 + 6];
    float s4 = 1.0f / (1.0f + __expf(-o4));
    float s5 = 1.0f / (1.0f + __expf(-o5));
    float s6 = 1.0f / (1.0f + __expf(-o6));
    float p  = sqrtf(s5 * s4) * s6;
    float cl = -fmaxf(__logf(p), -100.0f);
    float fgF = fg ? 1.0f : 0.0f;
    pk0[a] = make_float4(bx - 0.5f * bw, bx + 0.5f * bw,
                         by - 0.5f * bh, by + 0.5f * bh);
    pk1[a] = make_float4(bw * bh, cl + FG_PENALTY * (1.0f - fgF), fgF, dist);
    pk2[a] = make_float2(cx, cy);
}

// ---------------- kernel 2: per-GT streaming top-k ----------------------
__global__ void __launch_bounds__(256)
kPerGtTopK(const float4* __restrict__ pk0, const float4* __restrict__ pk1,
           const float2* __restrict__ pk2, const float* __restrict__ tgt,
           float* __restrict__ thrC, int* __restrict__ thrI, int A) {
    __shared__ float sIou[256 * NCAND];
    __shared__ float sC[256 * NCAND];
    __shared__ int   sI[256 * NCAND];
    const int g = blockIdx.x, tid = threadIdx.x;
    const float gx = tgt[g * 5 + 1], gy = tgt[g * 5 + 2];
    const float gw = tgt[g * 5 + 3], gh = tgt[g * 5 + 4];
    GtConst gt;
    gt.gxl = gx - 0.5f * gw; gt.gxr = gx + 0.5f * gw;
    gt.gyt = gy - 0.5f * gh; gt.gyb = gy + 0.5f * gh;
    gt.gx = gx; gt.gy = gy; gt.areaG = gw * gh;

    float iou10[NCAND], c10[NCAND]; int i10[NCAND];
#pragma unroll
    for (int i = 0; i < NCAND; ++i) { iou10[i] = -1.0f; c10[i] = INFINITY; i10[i] = 0x7fffffff; }

    for (int a = tid; a < A; a += blockDim.x) {
        __builtin_prefetch(&pk0[a + blockDim.x]);   // global_prefetch_b8
        AnchorConst an = loadAnchorPk(pk0, pk1, pk2, a);
        float iou, cost;
        eval_pair(an, gt, iou, cost);
        insIou(iou10, iou);
        insCost(c10, i10, cost, a);
    }
#pragma unroll
    for (int i = 0; i < NCAND; ++i) {
        sIou[tid * NCAND + i] = iou10[i];
        sC[tid * NCAND + i]   = c10[i];
        sI[tid * NCAND + i]   = i10[i];
    }
    __syncthreads();
    // Stage 1: 32 lanes each merge 8 thread lists into slot (lane*8).
    if (tid < 32) {
        int base = tid * 8;
        float mi[NCAND], mc[NCAND]; int mx[NCAND];
#pragma unroll
        for (int i = 0; i < NCAND; ++i) {
            mi[i] = sIou[base * NCAND + i];
            mc[i] = sC[base * NCAND + i];
            mx[i] = sI[base * NCAND + i];
        }
        for (int t = 1; t < 8; ++t)
            for (int i = 0; i < NCAND; ++i) {
                insIou(mi, sIou[(base + t) * NCAND + i]);
                insCost(mc, mx, sC[(base + t) * NCAND + i], sI[(base + t) * NCAND + i]);
            }
#pragma unroll
        for (int i = 0; i < NCAND; ++i) {
            sIou[base * NCAND + i] = mi[i];
            sC[base * NCAND + i]   = mc[i];
            sI[base * NCAND + i]   = mx[i];
        }
    }
    __syncthreads();
    // Stage 2: one lane merges the 32 partial lists, derives k and threshold.
    if (tid == 0) {
        float mi[NCAND], mc[NCAND]; int mx[NCAND];
#pragma unroll
        for (int i = 0; i < NCAND; ++i) { mi[i] = sIou[i]; mc[i] = sC[i]; mx[i] = sI[i]; }
        for (int b = 8; b < 256; b += 8)
            for (int i = 0; i < NCAND; ++i) {
                insIou(mi, sIou[b * NCAND + i]);
                insCost(mc, mx, sC[b * NCAND + i], sI[b * NCAND + i]);
            }
        float s = 0.0f;
#pragma unroll
        for (int i = 0; i < NCAND; ++i) s += mi[i];
        int k = (int)s;                 // trunc, matches astype(int32)
        if (k < 1) k = 1;
        if (k > NCAND) k = NCAND;
        thrC[g] = mc[k - 1];
        thrI[g] = mx[k - 1];
    }
}

// ---------------- kernel 3: per-anchor finalize (WMMA column sums) ------
__global__ void __launch_bounds__(256)
kFinalize(const float4* __restrict__ pk0, const float4* __restrict__ pk1,
          const float2* __restrict__ pk2, const float* __restrict__ tgt,
          const float* __restrict__ thrC, const int* __restrict__ thrI,
          float* __restrict__ fgOut, float* __restrict__ predOut,
          int* __restrict__ mgOut, int* __restrict__ numFg, int A, int G) {
    __shared__ float4 sBox[MAXG];    // gxl, gxr, gyt, gyb     (ds_load_b128)
    __shared__ float4 sMeta[MAXG];   // gx, gy, areaG, thrC    (ds_load_b128)
    __shared__ int    sTI[MAXG];     // threshold anchor index (ds_load_b32)
    const int tid = threadIdx.x;
    for (int i = tid; i < G; i += blockDim.x) {
        float gx = tgt[i * 5 + 1], gy = tgt[i * 5 + 2];
        float gw = tgt[i * 5 + 3], gh = tgt[i * 5 + 4];
        sBox[i]  = make_float4(gx - 0.5f * gw, gx + 0.5f * gw,
                               gy - 0.5f * gh, gy + 0.5f * gh);
        sMeta[i] = make_float4(gx, gy, gw * gh, thrC[i]);
        sTI[i]   = thrI[i];
    }
    __syncthreads();

    const int wave = tid >> 5;
    const int lane = tid & 31;
    const int n    = lane & 15;       // anchor slot in 16-wide tile (B N-dim)
    const int kh   = lane >> 4;       // K-half of the 32-deep WMMA chunk
    const int a    = blockIdx.x * 128 + wave * 16 + n;
    const int ac   = (a < A) ? a : (A - 1);    // clamped loads keep EXEC full

    AnchorConst an = loadAnchorPk(pk0, pk1, pk2, ac);
    const int fgb = an.fg > 0.5f;

    V16H ones;
#pragma unroll
    for (int j = 0; j < 16; ++j) ones.e[j] = (_Float16)1.0f;

    v8f accM = {};   // D[m][n] = sum_g matching[g, anchor n]
    v8f accI = {};   // D[m][n] = sum_g matching*iou
    float minc = INFINITY;
    int   ming = 0, firstg = 0x7fffffff;

    // Full 32-deep chunks: no per-element guard, EXEC stays all-ones.
    const int nFull = G >> 5;
    for (int c = 0; c < nFull; ++c) {
        V16H bm, bi;
        const int gb = c * 32 + kh * 16;
#pragma unroll
        for (int j = 0; j < 16; ++j) {
            const int g = gb + j;
            float4 box  = sBox[g];
            float4 meta = sMeta[g];
            const int   ti = sTI[g];
            const float tc = meta.w;
            GtConst gt = {box.x, box.y, box.z, box.w, meta.x, meta.y, meta.z};
            float iou, cost;
            eval_pair(an, gt, iou, cost);
            // rank < dynamic_k  <=>  (cost, idx) <=_lex threshold entry;
            // idx == thrI rescues bitwise cost mismatch at the pivot.
            int mm = fgb & ((a == ti) | (cost < tc) | ((cost == tc) & (a < ti)));
            bm.e[j] = (_Float16)(mm ? 1.0f : 0.0f);
            bi.e[j] = (_Float16)(mm ? iou : 0.0f);
            if (cost < minc) { minc = cost; ming = g; }
            firstg = (mm & (g < firstg)) ? g : firstg;
        }
        accM = __builtin_amdgcn_wmma_f32_16x16x32_f16(false, ones.v, false, bm.v,
                                                      (short)0, accM, false, false);
        accI = __builtin_amdgcn_wmma_f32_16x16x32_f16(false, ones.v, false, bi.v,
                                                      (short)0, accI, false, false);
    }
    // Masked tail chunk (only when G is not a multiple of 32).
    if (G & 31) {
        V16H bm, bi;
        const int gb = nFull * 32 + kh * 16;
#pragma unroll
        for (int j = 0; j < 16; ++j) {
            const int g = gb + j;
            const int valid = g < G;
            const int gc = valid ? g : (G - 1);
            float4 box  = sBox[gc];
            float4 meta = sMeta[gc];
            const int   ti = sTI[gc];
            const float tc = meta.w;
            GtConst gt = {box.x, box.y, box.z, box.w, meta.x, meta.y, meta.z};
            float iou, cost;
            eval_pair(an, gt, iou, cost);
            int mm = valid & fgb & ((a == ti) | (cost < tc) | ((cost == tc) & (a < ti)));
            bm.e[j] = (_Float16)(mm ? 1.0f : 0.0f);
            bi.e[j] = (_Float16)(mm ? iou : 0.0f);
            if (valid && cost < minc) { minc = cost; ming = g; }
            firstg = (mm & (g < firstg)) ? g : firstg;
        }
        accM = __builtin_amdgcn_wmma_f32_16x16x32_f16(false, ones.v, false, bm.v,
                                                      (short)0, accM, false, false);
        accI = __builtin_amdgcn_wmma_f32_16x16x32_f16(false, ones.v, false, bi.v,
                                                      (short)0, accI, false, false);
    }

    // Combine the two K-halves' scalar trackers (lanes l <-> l^16).
    float ominc = __shfl_xor(minc, 16, 32);
    int   oming = __shfl_xor(ming, 16, 32);
    if (ominc < minc || (ominc == minc && oming < ming)) {
        minc = ominc; ming = oming;
    }
    int ofirst = __shfl_xor(firstg, 16, 32);
    firstg = min(firstg, ofirst);

    if (lane < 16 && a < A) {
        const float cntF   = accM[0];   // lane n holds D[0][n]
        const float iouSum = accI[0];
        const int cnt = (int)(cntF + 0.5f);
        float fgF, pred; int mg;
        if (cnt > 1) {
            // multi-matched -> onehot(argmin cost): recompute IoU at the winner
            // once instead of tracking it through the whole scan.
            float4 box  = sBox[ming];
            float4 meta = sMeta[ming];
            GtConst gt = {box.x, box.y, box.z, box.w, meta.x, meta.y, meta.z};
            float iou, cost;
            eval_pair(an, gt, iou, cost);
            fgF = 1.0f; pred = iou; mg = ming;
        } else if (cnt == 1) {
            fgF = 1.0f; pred = iouSum; mg = firstg;
        } else {
            fgF = 0.0f; pred = 0.0f;   mg = -1;
        }
        fgOut[a]   = fgF;
        predOut[a] = pred;
        mgOut[a]   = mg;
        if (fgF > 0.5f) atomicAdd(numFg, 1);
    }
}

extern "C" void kernel_launch(void* const* d_in, const int* in_sizes, int n_in,
                              void* d_out, int out_size, void* d_ws, size_t ws_size,
                              hipStream_t stream) {
    const float* out7   = (const float*)d_in[0];
    const float* tgt    = (const float*)d_in[1];
    const float* grid   = (const float*)d_in[2];
    const float* stride = (const float*)d_in[3];
    const int A = in_sizes[3];
    const int G = in_sizes[1] / 5;

    char* wsb = (char*)d_ws;
    float4* pk0 = (float4*)wsb;
    float4* pk1 = (float4*)(wsb + (size_t)16 * A);
    float2* pk2 = (float2*)(wsb + (size_t)32 * A);
    float*  thrC = (float*)(wsb + (size_t)40 * A);
    int*    thrI = (int*)(wsb + (size_t)40 * A + (size_t)4 * G);

    float* fgOut   = (float*)d_out;
    float* predOut = fgOut + A;
    int*   mgOut   = (int*)d_out + (size_t)2 * A;
    int*   tail    = (int*)d_out + (size_t)3 * A;  // [num_fg, G]

    kInit<<<1, 1, 0, stream>>>(tail, G);
    kAnchorPrep<<<(A + 255) / 256, 256, 0, stream>>>(out7, tgt, grid, stride,
                                                     pk0, pk1, pk2, A, G);
    kPerGtTopK<<<G, 256, 0, stream>>>(pk0, pk1, pk2, tgt, thrC, thrI, A);
    kFinalize<<<(A + 127) / 128, 256, 0, stream>>>(pk0, pk1, pk2, tgt, thrC, thrI,
                                                   fgOut, predOut, mgOut, tail, A, G);
}
